// GraphTripleConv_67980742361871
// MI455X (gfx1250) — matrix operand order repaired and checked by reading, not compiled, not run on hardware
//
#include <hip/hip_runtime.h>

// ---------------------------------------------------------------------------
// GraphTripleConv on MI455X (gfx1250): f16-input / f32-accumulate WMMA GEMMs.
// ---------------------------------------------------------------------------

typedef _Float16 v16h __attribute__((ext_vector_type(16)));
typedef _Float16 v8h  __attribute__((ext_vector_type(8)));
typedef float    v8f  __attribute__((ext_vector_type(8)));
typedef float    v4f  __attribute__((ext_vector_type(4)));

#define O_N   50000
#define T_N   100000
#define DIN_N 512
#define H_N   512
#define DOUT_N 512
#define NOISE_N 2

// D = A(16x32 f16) x B(32x16 f16) + C(16x16 f32), one WMMA per K=32 step.
__device__ __forceinline__ v8f wmma_step(v16h a, v16h b, v8f c) {
    return __builtin_amdgcn_wmma_f32_16x16x32_f16(
        /*neg_a=*/false, a, /*neg_b=*/false, b,
        /*c_mod=*/(short)0, c, /*reuse_a=*/false, /*reuse_b=*/false);
}

union AFrag { v16h v; v8h h[2]; };

// ---------------------------------------------------------------------------
// Prep kernels
// ---------------------------------------------------------------------------

// obj = concat(embeddings[O x 510], noise[O x 2]) -> f16 [O x 512]
__global__ void build_obj_kernel(const float* __restrict__ emb,
                                 const float* __restrict__ noise,
                                 _Float16* __restrict__ objH) {
    int i = blockIdx.x * blockDim.x + threadIdx.x;
    if (i >= O_N * DIN_N) return;
    int row = i >> 9;
    int col = i & 511;
    float v = (col < (DIN_N - NOISE_N)) ? emb[(size_t)row * (DIN_N - NOISE_N) + col]
                                        : noise[(size_t)row * NOISE_N + (col - (DIN_N - NOISE_N))];
    objH[i] = (_Float16)v;
}

__global__ void f32_to_f16_kernel(const float* __restrict__ src,
                                  _Float16* __restrict__ dst, int n) {
    int i = blockIdx.x * blockDim.x + threadIdx.x;
    if (i < n) dst[i] = (_Float16)src[i];
}

__global__ void zero_f32_kernel(float* __restrict__ p, long long n) {
    long long i = (long long)blockIdx.x * blockDim.x + threadIdx.x;
    if (i < n) p[i] = 0.0f;
}

// ---------------------------------------------------------------------------
// Edge kernel: one wave per 16-edge tile.
//   t_vecs = concat(obj[s], obj[o]) * w         (16 x 1024, LDS f16)
//   h1     = relu(t_vecs @ W1a^T + b1a)          (16 x 512,  LDS f16)
//   new_t  = relu(h1 @ W1b^T + b1b)              -> atomic scatter into pooled
// ---------------------------------------------------------------------------
__global__ __launch_bounds__(32)
void edge_kernel(const _Float16* __restrict__ objH,
                 const long long* __restrict__ edges,
                 const float* __restrict__ ew,
                 const _Float16* __restrict__ W1a,   // [512 x 1024] row-major f16
                 const float* __restrict__ b1a,
                 const _Float16* __restrict__ W1b,   // [1024 x 512] row-major f16
                 const float* __restrict__ b1b,
                 float* __restrict__ pooled,         // [O x 512]
                 float* __restrict__ counts)         // [O]
{
    __shared__ _Float16 tv[16][2 * DIN_N];   // 32 KB
    __shared__ _Float16 h1[16][H_N];         // 16 KB
    __shared__ int   sIdx[16];
    __shared__ int   oIdx[16];
    __shared__ float wgt[16];

    const int lane = threadIdx.x;
    const int l  = lane & 15;
    const int hi = lane >> 4;
    const long long e0 = (long long)blockIdx.x * 16;

    // Load indices/weights; lanes 0-15 handle subject, 16-31 handle object.
    {
        long long idx = edges[(e0 + l) * 2 + hi];
        if (hi == 0) { sIdx[l] = (int)idx; wgt[l] = ew[e0 + l]; }
        else         { oIdx[l] = (int)idx; }
        atomicAdd(&counts[(int)idx], 1.0f);   // degree accumulation
    }
    __syncthreads();

    // Stage t_vecs: lane (l,hi) copies row obj[s or o] scaled by w into LDS.
    {
        const int row = (hi == 0) ? sIdx[l] : oIdx[l];
        const float w = wgt[l];
        const v8h* __restrict__ src = (const v8h*)(objH + (size_t)row * DIN_N);
        v8h* dst = (v8h*)(&tv[l][hi * DIN_N]);
        for (int j = 0; j < DIN_N / 8; ++j) {
            v8h x = src[j];
            v8h y;
#pragma unroll
            for (int u = 0; u < 8; ++u) y[u] = (_Float16)((float)x[u] * w);
            dst[j] = y;
        }
    }
    __syncthreads();

    // ---- GEMM1: h1[16x512] = relu(tv[16x1024] @ W1a^T + b1a), K = 1024 ----
    for (int nt = 0; nt < H_N / 16; ++nt) {
        const int n = nt * 16 + l;
        const _Float16* __restrict__ brow = W1a + (size_t)n * (2 * DIN_N) + hi * 16;
        v8f acc = {};
        for (int kb = 0; kb < 2 * DIN_N; kb += 32) {
            __builtin_prefetch(brow + kb + 128, 0, 0);   // stream next weight chunk
            AFrag a;
            a.h[0] = *(const v8h*)(&tv[l][kb + hi * 16]);
            a.h[1] = *(const v8h*)(&tv[l][kb + hi * 16 + 8]);
            v16h b = *(const v16h*)(brow + kb);
            acc = wmma_step(a.v, b, acc);
        }
        const float bias = b1a[n];
#pragma unroll
        for (int v = 0; v < 8; ++v) {
            const int m = v + 8 * hi;
            h1[m][n] = (_Float16)fmaxf(acc[v] + bias, 0.0f);
        }
    }
    __syncthreads();

    // ---- GEMM2 + scatter: new_t[16x1024] = relu(h1 @ W1b^T + b1b) ----
    for (int nt = 0; nt < (2 * H_N) / 16; ++nt) {
        const int n = nt * 16 + l;
        const _Float16* __restrict__ brow = W1b + (size_t)n * H_N + hi * 16;
        v8f acc = {};
        for (int kb = 0; kb < H_N; kb += 32) {
            AFrag a;
            a.h[0] = *(const v8h*)(&h1[l][kb + hi * 16]);
            a.h[1] = *(const v8h*)(&h1[l][kb + hi * 16 + 8]);
            v16h b = *(const v16h*)(brow + kb);
            acc = wmma_step(a.v, b, acc);
        }
        const float bias = b1b[n];
        const bool isS = (n < H_N);
        const int col = isS ? n : (n - H_N);
#pragma unroll
        for (int v = 0; v < 8; ++v) {
            const int m = v + 8 * hi;
            const int obj = isS ? sIdx[m] : oIdx[m];
            const float val = fmaxf(acc[v] + bias, 0.0f);
            atomicAdd(&pooled[(size_t)obj * H_N + col], val);
        }
    }
}

// ---------------------------------------------------------------------------
// Object kernel: one wave per 16-object tile.
//   A0  = (pooled / clip(counts,1)) as f16
//   h2  = relu(A0 @ W2a^T + b2a)
//   out = relu(h2 @ W2b^T + b2b)
// ---------------------------------------------------------------------------
__global__ __launch_bounds__(32)
void obj_kernel(const float* __restrict__ pooled,
                const float* __restrict__ counts,
                const _Float16* __restrict__ W2a,   // [512 x 512] f16
                const float* __restrict__ b2a,
                const _Float16* __restrict__ W2b,   // [512 x 512] f16
                const float* __restrict__ b2b,
                float* __restrict__ out)            // [O x 512] f32
{
    __shared__ _Float16 A0[16][H_N];   // 16 KB
    __shared__ _Float16 H2[16][H_N];   // 16 KB

    const int lane = threadIdx.x;
    const int l  = lane & 15;
    const int hi = lane >> 4;
    const int r0 = blockIdx.x * 16;

    // Normalize + convert to f16: lane (l,hi) covers columns [hi*256, hi*256+256).
    {
        const int row = r0 + l;
        const float inv = 1.0f / fmaxf(counts[row], 1.0f);
        const v4f* __restrict__ src = (const v4f*)(pooled + (size_t)row * H_N + hi * 256);
        for (int j = 0; j < 64; ++j) {
            v4f p = src[j];
            const int c0 = hi * 256 + j * 4;
#pragma unroll
            for (int u = 0; u < 4; ++u) A0[l][c0 + u] = (_Float16)(p[u] * inv);
        }
    }
    __syncthreads();

    // ---- GEMM3: H2 = relu(A0 @ W2a^T + b2a), K = 512 ----
    for (int nt = 0; nt < H_N / 16; ++nt) {
        const int n = nt * 16 + l;
        const _Float16* __restrict__ brow = W2a + (size_t)n * H_N + hi * 16;
        v8f acc = {};
        for (int kb = 0; kb < H_N; kb += 32) {
            AFrag a;
            a.h[0] = *(const v8h*)(&A0[l][kb + hi * 16]);
            a.h[1] = *(const v8h*)(&A0[l][kb + hi * 16 + 8]);
            v16h b = *(const v16h*)(brow + kb);
            acc = wmma_step(a.v, b, acc);
        }
        const float bias = b2a[n];
#pragma unroll
        for (int v = 0; v < 8; ++v) {
            const int m = v + 8 * hi;
            H2[m][n] = (_Float16)fmaxf(acc[v] + bias, 0.0f);
        }
    }
    __syncthreads();

    // ---- GEMM4: out = relu(H2 @ W2b^T + b2b), K = 512, f32 stores ----
    for (int nt = 0; nt < DOUT_N / 16; ++nt) {
        const int n = nt * 16 + l;
        const _Float16* __restrict__ brow = W2b + (size_t)n * H_N + hi * 16;
        v8f acc = {};
        for (int kb = 0; kb < H_N; kb += 32) {
            AFrag a;
            a.h[0] = *(const v8h*)(&H2[l][kb + hi * 16]);
            a.h[1] = *(const v8h*)(&H2[l][kb + hi * 16 + 8]);
            v16h b = *(const v16h*)(brow + kb);
            acc = wmma_step(a.v, b, acc);
        }
        const float bias = b2b[n];
#pragma unroll
        for (int v = 0; v < 8; ++v) {
            const int m = v + 8 * hi;
            out[(size_t)(r0 + m) * DOUT_N + n] = fmaxf(acc[v] + bias, 0.0f);
        }
    }
}

// ---------------------------------------------------------------------------
// Launch
// ---------------------------------------------------------------------------
extern "C" void kernel_launch(void* const* d_in, const int* in_sizes, int n_in,
                              void* d_out, int out_size, void* d_ws, size_t ws_size,
                              hipStream_t stream) {
    (void)in_sizes; (void)n_in; (void)out_size; (void)ws_size;

    const float*     emb   = (const float*)d_in[0];      // [O x 510]
    const float*     ew    = (const float*)d_in[1];      // [T x 1]
    const long long* edges = (const long long*)d_in[2];  // [T x 2] int64
    const float*     noise = (const float*)d_in[3];      // [O x 2]
    const float*     W1a   = (const float*)d_in[4];      // [512 x 1024]
    const float*     b1a   = (const float*)d_in[5];
    const float*     W1b   = (const float*)d_in[6];      // [1024 x 512]
    const float*     b1b   = (const float*)d_in[7];
    const float*     W2a   = (const float*)d_in[8];      // [512 x 512]
    const float*     b2a   = (const float*)d_in[9];
    const float*     W2b   = (const float*)d_in[10];     // [512 x 512]
    const float*     b2b   = (const float*)d_in[11];
    float* out = (float*)d_out;

    // Workspace carve-up (all offsets 256B-aligned).
    char* ws = (char*)d_ws;
    size_t off = 0;
    _Float16* objH   = (_Float16*)(ws + off); off += (size_t)O_N * DIN_N * 2;        // 51.2 MB
    _Float16* W1a_h  = (_Float16*)(ws + off); off += (size_t)H_N * (2 * DIN_N) * 2;  // 1 MB
    _Float16* W1b_h  = (_Float16*)(ws + off); off += (size_t)(2 * H_N) * H_N * 2;    // 1 MB
    _Float16* W2a_h  = (_Float16*)(ws + off); off += (size_t)H_N * H_N * 2;          // 0.5 MB
    _Float16* W2b_h  = (_Float16*)(ws + off); off += (size_t)DOUT_N * H_N * 2;       // 0.5 MB
    float*    pooled = (float*)(ws + off);    off += (size_t)O_N * H_N * 4;          // 102.4 MB
    float*    counts = (float*)(ws + off);    off += (size_t)O_N * 4;                // 0.2 MB
    // counts immediately follows pooled -> zero both with one launch.

    const int TPB = 256;

    // 1) Build f16 object matrix (concat + convert).
    build_obj_kernel<<<(O_N * DIN_N + TPB - 1) / TPB, TPB, 0, stream>>>(emb, noise, objH);

    // 2) Convert weights to f16.
    {
        int n;
        n = H_N * 2 * DIN_N;  f32_to_f16_kernel<<<(n + TPB - 1) / TPB, TPB, 0, stream>>>(W1a, W1a_h, n);
        n = 2 * H_N * H_N;    f32_to_f16_kernel<<<(n + TPB - 1) / TPB, TPB, 0, stream>>>(W1b, W1b_h, n);
        n = H_N * H_N;        f32_to_f16_kernel<<<(n + TPB - 1) / TPB, TPB, 0, stream>>>(W2a, W2a_h, n);
        n = DOUT_N * H_N;     f32_to_f16_kernel<<<(n + TPB - 1) / TPB, TPB, 0, stream>>>(W2b, W2b_h, n);
    }

    // 3) Zero pooled + counts (contiguous).
    {
        long long n = (long long)O_N * H_N + O_N;
        zero_f32_kernel<<<(int)((n + TPB - 1) / TPB), TPB, 0, stream>>>(pooled, n);
    }

    // 4) Edge pipeline: gather -> GEMM1 -> GEMM2 -> atomic scatter.
    edge_kernel<<<T_N / 16, 32, 0, stream>>>(objH, edges, ew,
                                             W1a_h, b1a, W1b_h, b1b,
                                             pooled, counts);

    // 5) Object pipeline: normalize -> GEMM3 -> GEMM4 -> out.
    obj_kernel<<<O_N / 16, 32, 0, stream>>>(pooled, counts,
                                            W2a_h, b2a, W2b_h, b2b, out);
}
